// Decoder_42339787604509
// MI455X (gfx1250) — matrix operand order repaired and checked by reading, not compiled, not run on hardware
//
#include <hip/hip_runtime.h>

// ---------------------------------------------------------------------------
// MI455X (gfx1250) decoder: 24-step LSTM scan with attention.
//  * all dense layers -> bf16 WMMA (v_wmma_f32_16x16x32_bf16), f32 accum
//  * K-loop software-pipelined (double-buffered fragments, K templated so the
//    buffers stay in registers) -> loads of step k+1 in flight across WMMAs
//    of step k instead of s_wait_loadcnt 0 before every WMMA
//  * input_encoded converted to bf16 once (151 MB -> fits 192 MB L2); the
//    per-step batched matvec (context) streams from L2 with global_prefetch
//  * gates for both LSTM directions fused into one GEMM 256x2048x1088 with a
//    block-structured packed weight (zero padding), K multiple of 32
// ---------------------------------------------------------------------------

typedef __bf16 bf16;
typedef __attribute__((ext_vector_type(16))) __bf16 v16bf;
typedef __attribute__((ext_vector_type(8)))  float  v8f;

#define HID   256
#define TLEN  576
#define FT    24
#define MF    32
#define BATCH 256
#define EDIM  512           // 2*HID
#define XK    544           // EDIM + MF (attn / W_ih input width)
#define GK    1088          // packed gates K: 544 + 256 + 256 + 32 pad (34*32)
#define GN    2048          // 2 directions * 4*HID
#define EPSD  1e-6f

__device__ __forceinline__ float sigm(float x) { return 1.0f / (1.0f + __expf(-x)); }

// ------------------------------ conversions --------------------------------
__global__ void cvt_f32_to_bf16(const float* __restrict__ in, bf16* __restrict__ out, long n) {
    long i = (long)blockIdx.x * blockDim.x + threadIdx.x;
    long stride = (long)gridDim.x * blockDim.x;
    for (; i < n; i += stride) out[i] = (bf16)in[i];
}

// ------------------------- packed gates weight -----------------------------
// Wg[row, col], row = d*1024 + g (g in torch i,f,g,o order), col layout:
//   [0,544)   : W_ih[d][g, :]            (input = [ctx | ff_t])
//   [544,800) : W_hh[0][g, :] if d==0 else 0
//   [800,1056): W_hh[1][g, :] if d==1 else 0
//   [1056,1088): 0 pad
__global__ void pack_wg(const float* __restrict__ Wih, const float* __restrict__ Whh,
                        bf16* __restrict__ Wg) {
    long idx = (long)blockIdx.x * blockDim.x + threadIdx.x;
    if (idx >= (long)GN * GK) return;
    int row = (int)(idx / GK), col = (int)(idx % GK);
    int d = row >> 10, g = row & 1023;
    float v = 0.0f;
    if (col < XK) {
        v = Wih[((long)d * 1024 + g) * XK + col];
    } else {
        int hb = XK + d * HID;
        if (col >= hb && col < hb + HID) v = Whh[((long)d * 1024 + g) * HID + (col - hb)];
    }
    Wg[idx] = (bf16)v;
}

__global__ void make_bias_g(const float* __restrict__ bih, const float* __restrict__ bhh,
                            float* __restrict__ bg) {
    int i = blockIdx.x * blockDim.x + threadIdx.x;
    if (i < GN) bg[i] = bih[i] + bhh[i];
}

// ------------------------------ alpha (once) -------------------------------
// dist[b,j,p] = sqrt(sum_m (fut[b,j,m] - hist[b,p*24+j,m] + EPS)^2)
// alpha[b,:] = softmax_j( 1 / sum_p dist[b,j,p] )
__global__ void alpha_kernel(const float* __restrict__ fut, const float* __restrict__ hist,
                             float* __restrict__ alpha) {
    int b = blockIdx.x;
    int j = threadIdx.x;                  // wave32; j<24 active
    __shared__ float dp[24];
    if (j < 24) {
        const float* fj = fut + ((long)b * FT + j) * MF;
        float sum = 0.0f;
        for (int p = 0; p < 24; ++p) {
            const float* hp = hist + ((long)b * TLEN + p * 24 + j) * MF;
            float ss = 0.0f;
            for (int m = 0; m < MF; ++m) { float d = fj[m] - hp[m] + EPSD; ss += d * d; }
            sum += sqrtf(ss);
        }
        dp[j] = 1.0f / sum;
    }
    __syncthreads();
    if (j < 24) {
        float mx = -1e30f;
        for (int k = 0; k < 24; ++k) mx = fmaxf(mx, dp[k]);
        float s = 0.0f;
        for (int k = 0; k < 24; ++k) s += __expf(dp[k] - mx);
        alpha[b * 24 + j] = __expf(dp[j] - mx) / s;
    }
}

// ------------------------------ WMMA GEMM ----------------------------------
// D[m,n] = act( sum_k A[m,k]*W[n,k] + bias[n] )
// A: [Mrows x K] bf16 row-major, W: [N x K] bf16 row-major. K%32==0, N%64==0.
// One wave computes a 16x64 tile. K templated -> full unroll, double-buffered
// fragments stay in registers, next-step loads overlap current WMMAs.
// EPI: 0 = store f32, 1 = tanh -> store bf16, 2 = relu -> store f32
template <int EPI, int K>
__global__ void gemm_bf16_wmma(const bf16* __restrict__ A, const bf16* __restrict__ W,
                               const float* __restrict__ bias, void* __restrict__ out,
                               int N) {
    const int lane   = threadIdx.x;      // block = 32 (one wave, EXEC all ones)
    const int r      = lane & 15;
    const int hi     = lane >> 4;
    const int n_base = blockIdx.x * 64;
    const int m_base = blockIdx.y * 16;

    v8f zero = {0.f, 0.f, 0.f, 0.f, 0.f, 0.f, 0.f, 0.f};
    v8f acc[4] = {zero, zero, zero, zero};

    // A 16x32 fragment: lane holds 16 contiguous K starting at 16*hi of row r
    const bf16* Arow = A + (long)(m_base + r) * K + 16 * hi;
    // B 32x16 fragment: B[k,n] = W[n,k]; lane holds 16 contiguous K starting at
    // 16*hi of weight row (n_base + j*16 + r)
    const bf16* Wr0 = W + (long)(n_base +  0 + r) * K + 16 * hi;
    const bf16* Wr1 = W + (long)(n_base + 16 + r) * K + 16 * hi;
    const bf16* Wr2 = W + (long)(n_base + 32 + r) * K + 16 * hi;
    const bf16* Wr3 = W + (long)(n_base + 48 + r) * K + 16 * hi;

    union frag { v16bf v; uint4 u[2]; };
    frag af[2], wf[2][4];

    // prologue: load k0 = 0 fragments
    af[0].u[0] = *reinterpret_cast<const uint4*>(Arow);
    af[0].u[1] = *reinterpret_cast<const uint4*>(Arow + 8);
    wf[0][0].u[0] = *reinterpret_cast<const uint4*>(Wr0);
    wf[0][0].u[1] = *reinterpret_cast<const uint4*>(Wr0 + 8);
    wf[0][1].u[0] = *reinterpret_cast<const uint4*>(Wr1);
    wf[0][1].u[1] = *reinterpret_cast<const uint4*>(Wr1 + 8);
    wf[0][2].u[0] = *reinterpret_cast<const uint4*>(Wr2);
    wf[0][2].u[1] = *reinterpret_cast<const uint4*>(Wr2 + 8);
    wf[0][3].u[0] = *reinterpret_cast<const uint4*>(Wr3);
    wf[0][3].u[1] = *reinterpret_cast<const uint4*>(Wr3 + 8);

#pragma unroll
    for (int k0 = 0; k0 < K; k0 += 32) {
        const int cur = (k0 / 32) & 1;
        const int nxt = cur ^ 1;
        if (k0 + 32 < K) {   // prefetch next K-step while current WMMAs run
            af[nxt].u[0] = *reinterpret_cast<const uint4*>(Arow + k0 + 32);
            af[nxt].u[1] = *reinterpret_cast<const uint4*>(Arow + k0 + 40);
            wf[nxt][0].u[0] = *reinterpret_cast<const uint4*>(Wr0 + k0 + 32);
            wf[nxt][0].u[1] = *reinterpret_cast<const uint4*>(Wr0 + k0 + 40);
            wf[nxt][1].u[0] = *reinterpret_cast<const uint4*>(Wr1 + k0 + 32);
            wf[nxt][1].u[1] = *reinterpret_cast<const uint4*>(Wr1 + k0 + 40);
            wf[nxt][2].u[0] = *reinterpret_cast<const uint4*>(Wr2 + k0 + 32);
            wf[nxt][2].u[1] = *reinterpret_cast<const uint4*>(Wr2 + k0 + 40);
            wf[nxt][3].u[0] = *reinterpret_cast<const uint4*>(Wr3 + k0 + 32);
            wf[nxt][3].u[1] = *reinterpret_cast<const uint4*>(Wr3 + k0 + 40);
        }
#pragma unroll
        for (int j = 0; j < 4; ++j)
            acc[j] = __builtin_amdgcn_wmma_f32_16x16x32_bf16(
                false, af[cur].v, false, wf[cur][j].v, (short)0, acc[j], false, false);
    }

#pragma unroll
    for (int j = 0; j < 4; ++j) {
        int col = n_base + j * 16 + r;
        float bv = bias ? bias[col] : 0.0f;
#pragma unroll
        for (int v = 0; v < 8; ++v) {
            int row = m_base + v + 8 * hi;      // C/D layout: m = vgpr + 8*(lane/16)
            float x = acc[j][v] + bv;
            if (EPI == 1)      ((bf16*)out)[(long)row * N + col] = (bf16)tanhf(x);
            else if (EPI == 2) ((float*)out)[(long)row * N + col] = fmaxf(x, 0.0f);
            else               ((float*)out)[(long)row * N + col] = x;
        }
    }
}

// --------------------------- per-step elementwise --------------------------
__global__ void pack_x_attn(const float* __restrict__ h, const float* __restrict__ fut,
                            bf16* __restrict__ x, int tstep) {
    int idx = blockIdx.x * blockDim.x + threadIdx.x;
    if (idx >= BATCH * XK) return;
    int b = idx / XK, col = idx % XK;
    float v;
    if (col < HID)            v = h[(long)b * HID + col];                           // h[0]
    else if (col < 2 * HID)   v = h[(long)BATCH * HID + (long)b * HID + col - HID]; // h[1]
    else                      v = fut[((long)b * FT + tstep) * MF + (col - 2 * HID)];
    x[idx] = (bf16)v;
}

__global__ void softmax_alpha_kernel(const float* __restrict__ a2,
                                     const float* __restrict__ alpha,
                                     float* __restrict__ s) {
    int b = blockIdx.x, tid = threadIdx.x;     // block = 64
    const float* row = a2 + (long)b * TLEN;
    __shared__ float lds[64];
    float mx = -1e30f;
    for (int i = tid; i < TLEN; i += 64) mx = fmaxf(mx, row[i]);
    lds[tid] = mx; __syncthreads();
    for (int o = 32; o > 0; o >>= 1) { if (tid < o) lds[tid] = fmaxf(lds[tid], lds[tid + o]); __syncthreads(); }
    mx = lds[0]; __syncthreads();
    float sm = 0.0f;
    for (int i = tid; i < TLEN; i += 64) sm += __expf(row[i] - mx);
    lds[tid] = sm; __syncthreads();
    for (int o = 32; o > 0; o >>= 1) { if (tid < o) lds[tid] += lds[tid + o]; __syncthreads(); }
    float inv = 1.0f / lds[0];
    const float* al = alpha + b * 24;
    for (int i = tid; i < TLEN; i += 64)
        s[(long)b * TLEN + i] = __expf(row[i] - mx) * inv * al[i / 24];
}

// ctx[b,e] = sum_t s[b,t] * enc_bf16[b,t,e]   (batched matvec; enc streams from
// L2; global_prefetch_b8 issued 64 rows ahead per 16-row chunk)
__global__ void context_kernel(const float* __restrict__ s, const bf16* __restrict__ enc,
                               float* __restrict__ ctx) {
    int b = blockIdx.y;
    int e = blockIdx.x * 256 + threadIdx.x;
    __shared__ float sw[TLEN];
    for (int i = threadIdx.x; i < TLEN; i += 256) sw[i] = s[(long)b * TLEN + i];
    __syncthreads();
    const bf16* encb = enc + (long)b * TLEN * EDIM + e;
    float acc = 0.0f;
    for (int t0 = 0; t0 < TLEN; t0 += 16) {
        // speculative prefetch (dropped if OOB); 64 rows = 64 KB ahead
        __builtin_prefetch(encb + (long)(t0 + 64) * EDIM, 0, 0);
#pragma unroll
        for (int t = t0; t < t0 + 16; ++t)
            acc = fmaf(sw[t], (float)encb[(long)t * EDIM], acc);
    }
    ctx[(long)b * EDIM + e] = acc;
}

__global__ void pack_xg(const float* __restrict__ ctx, const float* __restrict__ fut,
                        const float* __restrict__ h, bf16* __restrict__ xg, int tstep) {
    int idx = blockIdx.x * blockDim.x + threadIdx.x;
    if (idx >= BATCH * GK) return;
    int b = idx / GK, col = idx % GK;
    float v = 0.0f;
    if (col < EDIM)              v = ctx[(long)b * EDIM + col];
    else if (col < XK)           v = fut[((long)b * FT + tstep) * MF + (col - EDIM)];
    else if (col < XK + HID)     v = h[(long)b * HID + (col - XK)];
    else if (col < XK + 2 * HID) v = h[(long)BATCH * HID + (long)b * HID + (col - XK - HID)];
    xg[idx] = (bf16)v;
}

__global__ void lstm_elem(const float* __restrict__ gates, float* __restrict__ h,
                          float* __restrict__ c, bf16* __restrict__ hh) {
    int idx = blockIdx.x * blockDim.x + threadIdx.x;   // over 2*B*H
    if (idx >= 2 * BATCH * HID) return;
    int d = idx / (BATCH * HID);
    int rem = idx % (BATCH * HID);
    int b = rem / HID, k = rem % HID;
    const float* g = gates + (long)b * GN + d * 1024;  // i,f,g,o blocks of 256
    float ig = g[k], fg = g[256 + k], gg = g[512 + k], og = g[768 + k];
    float cn = sigm(fg) * c[idx] + sigm(ig) * tanhf(gg);
    float hn = sigm(og) * tanhf(cn);
    c[idx] = cn;
    h[idx] = hn;
    hh[(long)b * (2 * HID) + d * HID + k] = (bf16)hn;  // [h0|h1] for fc_final
}

__global__ void fc2_kernel(const float* __restrict__ fc1o, const float* __restrict__ w2,
                           const float* __restrict__ b2, float* __restrict__ out, int tstep) {
    int b = blockIdx.x, tid = threadIdx.x;   // block = 256
    __shared__ float lds[256];
    lds[tid] = fc1o[(long)b * HID + tid] * w2[tid];   // fc1o already ReLU'd
    __syncthreads();
    for (int o = 128; o > 0; o >>= 1) { if (tid < o) lds[tid] += lds[tid + o]; __syncthreads(); }
    if (tid == 0) out[b * FT + tstep] = lds[0] + b2[0];
}

// ------------------------------ host driver --------------------------------
extern "C" void kernel_launch(void* const* d_in, const int* in_sizes, int n_in,
                              void* d_out, int out_size, void* d_ws, size_t ws_size,
                              hipStream_t stream) {
    (void)in_sizes; (void)n_in; (void)out_size; (void)ws_size;

    const float* enc_f = (const float*)d_in[0];
    const float* hist  = (const float*)d_in[1];
    const float* fut   = (const float*)d_in[3];
    const float* h0in  = (const float*)d_in[4];
    const float* c0in  = (const float*)d_in[5];
    const float* W1 = (const float*)d_in[6];  const float* b1 = (const float*)d_in[7];
    const float* W2 = (const float*)d_in[8];  const float* b2 = (const float*)d_in[9];
    const float* Wih = (const float*)d_in[10]; const float* Whh = (const float*)d_in[11];
    const float* bih = (const float*)d_in[12]; const float* bhh = (const float*)d_in[13];
    const float* fW1 = (const float*)d_in[14]; const float* fb1 = (const float*)d_in[15];
    const float* fW2 = (const float*)d_in[16]; const float* fb2 = (const float*)d_in[17];
    float* out = (float*)d_out;

    // workspace carve-out (~164 MB; dominated by bf16 copy of input_encoded)
    char* base = (char*)d_ws; size_t off = 0;
    auto alloc = [&](size_t bytes) -> void* {
        off = (off + 255) & ~(size_t)255;
        void* r = base + off; off += bytes; return r;
    };
    bf16*  enc_b  = (bf16*)alloc((size_t)BATCH * TLEN * EDIM * 2);
    bf16*  W1b    = (bf16*)alloc((size_t)TLEN * XK * 2);
    bf16*  W2b    = (bf16*)alloc((size_t)TLEN * TLEN * 2);
    bf16*  Wgb    = (bf16*)alloc((size_t)GN * GK * 2);
    bf16*  fW1b   = (bf16*)alloc((size_t)HID * EDIM * 2);
    float* bg     = (float*)alloc((size_t)GN * 4);
    float* alpha  = (float*)alloc((size_t)BATCH * 24 * 4);
    bf16*  xbuf   = (bf16*)alloc((size_t)BATCH * XK * 2);
    bf16*  a1     = (bf16*)alloc((size_t)BATCH * TLEN * 2);
    float* a2     = (float*)alloc((size_t)BATCH * TLEN * 4);
    float* sbuf   = (float*)alloc((size_t)BATCH * TLEN * 4);
    float* ctx    = (float*)alloc((size_t)BATCH * EDIM * 4);
    bf16*  xg     = (bf16*)alloc((size_t)BATCH * GK * 2);
    float* gates  = (float*)alloc((size_t)BATCH * GN * 4);
    float* hbuf   = (float*)alloc((size_t)2 * BATCH * HID * 4);
    float* cbuf   = (float*)alloc((size_t)2 * BATCH * HID * 4);
    bf16*  hh     = (bf16*)alloc((size_t)BATCH * 2 * HID * 2);
    float* fc1o   = (float*)alloc((size_t)BATCH * HID * 4);

    auto cvt = [&](const float* src, bf16* dst, long n) {
        long blocks = (n + 255) / 256; if (blocks > 32768) blocks = 32768;
        cvt_f32_to_bf16<<<(int)blocks, 256, 0, stream>>>(src, dst, n);
    };

    // --- one-time prep ---
    cvt(enc_f, enc_b, (long)BATCH * TLEN * EDIM);
    cvt(W1, W1b, (long)TLEN * XK);
    cvt(W2, W2b, (long)TLEN * TLEN);
    cvt(fW1, fW1b, (long)HID * EDIM);
    pack_wg<<<((long)GN * GK + 255) / 256, 256, 0, stream>>>(Wih, Whh, Wgb);
    make_bias_g<<<GN / 256, 256, 0, stream>>>(bih, bhh, bg);
    alpha_kernel<<<BATCH, 32, 0, stream>>>(fut, hist, alpha);
    hipMemcpyAsync(hbuf, h0in, (size_t)2 * BATCH * HID * 4, hipMemcpyDeviceToDevice, stream);
    hipMemcpyAsync(cbuf, c0in, (size_t)2 * BATCH * HID * 4, hipMemcpyDeviceToDevice, stream);

    // --- sequential scan over future steps ---
    for (int t = 0; t < FT; ++t) {
        pack_x_attn<<<(BATCH * XK + 255) / 256, 256, 0, stream>>>(hbuf, fut, xbuf, t);
        // attn1: [256 x 576 x 544], tanh -> bf16
        gemm_bf16_wmma<1, XK><<<dim3(TLEN / 64, BATCH / 16), 32, 0, stream>>>(xbuf, W1b, b1, a1, TLEN);
        // attn2: [256 x 576 x 576] -> f32
        gemm_bf16_wmma<0, TLEN><<<dim3(TLEN / 64, BATCH / 16), 32, 0, stream>>>(a1, W2b, b2, a2, TLEN);
        softmax_alpha_kernel<<<BATCH, 64, 0, stream>>>(a2, alpha, sbuf);
        context_kernel<<<dim3(EDIM / 256, BATCH), 256, 0, stream>>>(sbuf, enc_b, ctx);
        pack_xg<<<(BATCH * GK + 255) / 256, 256, 0, stream>>>(ctx, fut, hbuf, xg, t);
        // fused gates (both directions + recurrent): [256 x 2048 x 1088] -> f32
        gemm_bf16_wmma<0, GK><<<dim3(GN / 64, BATCH / 16), 32, 0, stream>>>(xg, Wgb, bg, gates, GN);
        lstm_elem<<<(2 * BATCH * HID + 255) / 256, 256, 0, stream>>>(gates, hbuf, cbuf, hh);
        // fc1: [256 x 256 x 512], relu -> f32
        gemm_bf16_wmma<2, EDIM><<<dim3(HID / 64, BATCH / 16), 32, 0, stream>>>(hh, fW1b, fb1, fc1o, HID);
        fc2_kernel<<<BATCH, 256, 0, stream>>>(fc1o, fW2, fb2, out, t);
    }
}
